// SimpleRGAT_26723286515871
// MI455X (gfx1250) — matrix operand order repaired and checked by reading, not compiled, not run on hardware
//
#include <hip/hip_runtime.h>
#include <hip/hip_bf16.h>

typedef __attribute__((ext_vector_type(16))) __bf16 v16bf;
typedef __attribute__((ext_vector_type(8)))  float  v8f;

#define HDIM 128

__device__ __forceinline__ unsigned short f2bf(float f) {
  unsigned int u = __builtin_bit_cast(unsigned int, f);
  u += 0x7FFFu + ((u >> 16) & 1u);               // round-to-nearest-even
  return (unsigned short)(u >> 16);
}

union Frag16 { unsigned int u[8]; v16bf v; };

// ---------------------------------------------------------------------------
// C[M,128] = act(A[M,128] @ B[128,128] + bias)   (bf16 WMMA, f32 accumulate)
// A is f32 or bf16 (ABF16); outputs optionally f32 (C) and/or bf16 (Cb).
// Block: 256 threads (8 waves), 64-row tile. Wave w: row-tile rt=w>>1 (16 rows),
// column half ch=w&1 (64 cols = 4 WMMA col tiles), 4 k-steps of 32.
// ---------------------------------------------------------------------------
template <bool RELU, bool ABF16, bool OUTF32, bool OUTBF16>
__global__ __launch_bounds__(256) void k_gemm64(const void* __restrict__ Av,
                                                const float* __restrict__ B,
                                                const float* __restrict__ bias,
                                                float* __restrict__ C,
                                                unsigned short* __restrict__ Cb,
                                                int M) {
  __shared__ __attribute__((aligned(16))) unsigned short As[64 * HDIM];   // [m][k]
  __shared__ __attribute__((aligned(16))) unsigned short Bs[HDIM * HDIM]; // [n][k]
  const int tid = threadIdx.x;
  const long m0 = (long)blockIdx.x * 64;

  if constexpr (ABF16) {
    const unsigned short* Ab = (const unsigned short*)Av;
    const int row = tid >> 2, q = tid & 3;        // 4 threads per row, 32 bf16 each
    const long gm = m0 + row;
    uint4 vals[4] = {};
    if (gm < M) {
      const uint4* sp = (const uint4*)(Ab + gm * HDIM + q * 32);
#pragma unroll
      for (int j = 0; j < 4; ++j) vals[j] = sp[j];
    }
    uint4* dp = (uint4*)(&As[row * HDIM + q * 32]);
#pragma unroll
    for (int j = 0; j < 4; ++j) dp[j] = vals[j];
  } else {
    const float* Af = (const float*)Av;
    for (int i = tid; i < 64 * HDIM; i += 256) {
      int r = i >> 7, k = i & 127;
      long gm = m0 + r;
      float v = (gm < M) ? Af[gm * HDIM + k] : 0.0f;
      As[i] = f2bf(v);
    }
  }
  for (int i = tid; i < HDIM * HDIM; i += 256) {  // stage B transposed [n][k]
    int k = i >> 7, n = i & 127;
    Bs[n * HDIM + k] = f2bf(B[i]);
  }
  __syncthreads();

  const int wave = tid >> 5, lane = tid & 31;
  const int rt = wave >> 1, ch = wave & 1;
  const int lm = lane & 15, lh = lane >> 4;

  v8f acc[4] = {};
#pragma unroll
  for (int kk = 0; kk < 4; ++kk) {
    const int kb = kk * 32;
    Frag16 a;
    const unsigned short* ar = &As[(rt * 16 + lm) * HDIM + kb + lh * 8];
#pragma unroll
    for (int j = 0; j < 4; ++j) {
      a.u[j]     = *(const unsigned int*)(ar + 2 * j);
      a.u[4 + j] = *(const unsigned int*)(ar + 16 + 2 * j);
    }
#pragma unroll
    for (int c = 0; c < 4; ++c) {
      Frag16 b;
      const unsigned short* br = &Bs[(ch * 64 + c * 16 + lm) * HDIM + kb + lh * 8];
#pragma unroll
      for (int j = 0; j < 4; ++j) {
        b.u[j]     = *(const unsigned int*)(br + 2 * j);
        b.u[4 + j] = *(const unsigned int*)(br + 16 + 2 * j);
      }
      acc[c] = __builtin_amdgcn_wmma_f32_16x16x32_bf16(false, a.v, false, b.v,
                                                       (short)0, acc[c], false, false);
    }
  }

#pragma unroll
  for (int c = 0; c < 4; ++c) {
    int gn = ch * 64 + c * 16 + lm;
    float bv = bias[gn];
#pragma unroll
    for (int i = 0; i < 8; ++i) {
      long gm = m0 + rt * 16 + lh * 8 + i;
      if (gm < M) {
        float v = acc[c][i] + bv;
        if (RELU) v = fmaxf(v, 0.0f);
        if (OUTF32) C[gm * HDIM + gn] = v;
        if (OUTBF16) Cb[gm * HDIM + gn] = f2bf(v);
      }
    }
  }
}

// ---------------------------------------------------------------------------
// W_rel f32 [r][k][n]  ->  bf16 transposed [r][n][k]  (one-time, tiny)
// ---------------------------------------------------------------------------
__global__ void k_convW(const float* __restrict__ Wrel,
                        unsigned short* __restrict__ WrelT, int total) {
  int i = blockIdx.x * blockDim.x + threadIdx.x;
  if (i < total) {
    int r = i >> 14, rem = i & 16383;
    int k = rem >> 7, n = rem & 127;
    WrelT[(r << 14) + n * HDIM + k] = f2bf(Wrel[i]);
  }
}

// ---------------------------------------------------------------------------
// Edge sort (counting sort by relation, regions padded to 64 edges)
// rmeta: [0..15]=rcount  [16..31]=roff  [32..47]=rcursor
// ---------------------------------------------------------------------------
__global__ void k_hist(const int* __restrict__ et, const int* __restrict__ ei,
                       float* __restrict__ cnt, int* __restrict__ rmeta, int E) {
  int e = blockIdx.x * blockDim.x + threadIdx.x;
  if (e < E) {
    atomicAdd(&rmeta[et[e]], 1);
    unsafeAtomicAdd(&cnt[ei[E + e]], 1.0f);     // per-dst in-degree
  }
}

__global__ void k_scan(int* rmeta, int R) {
  if (blockIdx.x == 0 && threadIdx.x == 0) {
    int off = 0;
    for (int r = 0; r < R; ++r) {
      rmeta[16 + r] = off;
      rmeta[32 + r] = 0;
      off += (rmeta[r] + 63) & ~63;             // 64-edge alignment per relation
    }
  }
}

__global__ void k_scatter(const int* __restrict__ et, int* __restrict__ rmeta,
                          int* __restrict__ sorted, int E) {
  int e = blockIdx.x * blockDim.x + threadIdx.x;
  if (e < E) {
    int r = et[e];
    int p = atomicAdd(&rmeta[32 + r], 1);
    sorted[rmeta[16 + r] + p] = e;
  }
}

// ---------------------------------------------------------------------------
// Per-relation edge GEMM: 64-edge tile per block (uniform relation by
// construction). Gather bf16 h rows (vectorized b128), stage pre-transposed
// bf16 W_rel[r], WMMA, scatter-add f32 atomics into agg[dst].
// ---------------------------------------------------------------------------
__global__ __launch_bounds__(256) void k_edge_gemm(
    const int* __restrict__ sorted, const int* __restrict__ edge_index,
    const int* __restrict__ edge_type, const unsigned short* __restrict__ hb,
    const unsigned short* __restrict__ WrelT, const float* __restrict__ brel,
    float* __restrict__ agg, int Epad, int E) {
  __shared__ __attribute__((aligned(16))) unsigned short As[64 * HDIM];
  __shared__ __attribute__((aligned(16))) unsigned short Bs[HDIM * HDIM];
  __shared__ int s_eid[64];
  __shared__ int s_rel;
  const int tid = threadIdx.x;
  const long base = (long)blockIdx.x * 64;

  if (tid < 64) {
    long idx = base + tid;
    s_eid[tid] = (idx < Epad) ? sorted[idx] : -1;
  }
  __syncthreads();
  if (tid == 0) {
    int r = 0;
    for (int i = 0; i < 64; ++i) {
      int e = s_eid[i];
      if (e >= 0) { r = edge_type[e]; break; }
    }
    s_rel = r;
  }
  __syncthreads();
  const int rel = s_rel;

  {  // gather A rows (bf16, 4 threads x 32 elems per row, b128 vectorized)
    const int row = tid >> 2, q = tid & 3;
    const int e = s_eid[row];
    uint4 vals[4] = {};
    if (e >= 0) {
      const uint4* sp = (const uint4*)(hb + (long)edge_index[e] * HDIM + q * 32);
#pragma unroll
      for (int j = 0; j < 4; ++j) vals[j] = sp[j];
    }
    uint4* dp = (uint4*)(&As[row * HDIM + q * 32]);
#pragma unroll
    for (int j = 0; j < 4; ++j) dp[j] = vals[j];
  }
  {  // stage pre-transposed bf16 weights: straight b128 copy
    const uint4* sp = (const uint4*)(WrelT + ((long)rel << 14));
    uint4* dp = (uint4*)Bs;
#pragma unroll
    for (int j = 0; j < 8; ++j) dp[tid + j * 256] = sp[tid + j * 256];
  }
  __syncthreads();

  const int wave = tid >> 5, lane = tid & 31;
  const int rt = wave >> 1, ch = wave & 1;
  const int lm = lane & 15, lh = lane >> 4;

  v8f acc[4] = {};
#pragma unroll
  for (int kk = 0; kk < 4; ++kk) {
    const int kb = kk * 32;
    Frag16 a;
    const unsigned short* ar = &As[(rt * 16 + lm) * HDIM + kb + lh * 8];
#pragma unroll
    for (int j = 0; j < 4; ++j) {
      a.u[j]     = *(const unsigned int*)(ar + 2 * j);
      a.u[4 + j] = *(const unsigned int*)(ar + 16 + 2 * j);
    }
#pragma unroll
    for (int c = 0; c < 4; ++c) {
      Frag16 b;
      const unsigned short* br = &Bs[(ch * 64 + c * 16 + lm) * HDIM + kb + lh * 8];
#pragma unroll
      for (int j = 0; j < 4; ++j) {
        b.u[j]     = *(const unsigned int*)(br + 2 * j);
        b.u[4 + j] = *(const unsigned int*)(br + 16 + 2 * j);
      }
      acc[c] = __builtin_amdgcn_wmma_f32_16x16x32_bf16(false, a.v, false, b.v,
                                                       (short)0, acc[c], false, false);
    }
  }

#pragma unroll
  for (int c = 0; c < 4; ++c) {
    int gn = ch * 64 + c * 16 + lm;
    float bv = brel[rel * HDIM + gn];
#pragma unroll
    for (int i = 0; i < 8; ++i) {
      int row = rt * 16 + lh * 8 + i;
      int e = s_eid[row];
      if (e >= 0) {
        int dst = edge_index[E + e];
        unsafeAtomicAdd(&agg[(long)dst * HDIM + gn], acc[c][i] + bv);
      }
    }
  }
}

// h2 = h + agg / max(cnt,1)  ->  bf16 (overwrites hb; only bf16 GEMMs consume it)
__global__ void k_combine(const float* __restrict__ h, const float* __restrict__ agg,
                          const float* __restrict__ cnt,
                          unsigned short* __restrict__ h2b, int N) {
  long i = (long)blockIdx.x * blockDim.x + threadIdx.x;
  if (i < (long)N * HDIM) {
    float c = cnt[i >> 7];
    if (c < 1.0f) c = 1.0f;
    h2b[i] = f2bf(h[i] + agg[i] / c);
  }
}

// pred[i] = dot(emb[i,:], W_p) + b_p   (one wave32 per row)
__global__ __launch_bounds__(256) void k_pred(const float* __restrict__ emb,
                                              const float* __restrict__ Wp,
                                              const float* __restrict__ bp,
                                              float* __restrict__ pred, int N) {
  int row = blockIdx.x * 8 + (threadIdx.x >> 5);
  int lane = threadIdx.x & 31;
  if (row >= N) return;
  float s = 0.0f;
#pragma unroll
  for (int j = lane; j < HDIM; j += 32) s += emb[(long)row * HDIM + j] * Wp[j];
#pragma unroll
  for (int off = 16; off > 0; off >>= 1) s += __shfl_down(s, off, 32);
  if (lane == 0) pred[row] = s + bp[0];
}

// ---------------------------------------------------------------------------
extern "C" void kernel_launch(void* const* d_in, const int* in_sizes, int n_in,
                              void* d_out, int out_size, void* d_ws, size_t ws_size,
                              hipStream_t stream) {
  (void)n_in; (void)out_size; (void)ws_size;
  const float* x          = (const float*)d_in[0];
  const int*   edge_index = (const int*)d_in[1];
  const int*   edge_type  = (const int*)d_in[2];
  const float* W_in       = (const float*)d_in[3];
  const float* b_in       = (const float*)d_in[4];
  const float* W_rel      = (const float*)d_in[5];
  const float* b_rel      = (const float*)d_in[6];
  const float* W_o1       = (const float*)d_in[7];
  const float* b_o1       = (const float*)d_in[8];
  const float* W_o2       = (const float*)d_in[9];
  const float* b_o2       = (const float*)d_in[10];
  const float* W_p        = (const float*)d_in[11];
  const float* b_p        = (const float*)d_in[12];

  const int N = in_sizes[0] / HDIM;
  const int E = in_sizes[2];
  const int R = in_sizes[5] / (HDIM * HDIM);
  const long NH = (long)N * HDIM;
  const int Epad = E + 64 * R;

  char* ws = (char*)d_ws;
  float*          h      = (float*)ws;          ws += NH * sizeof(float);
  float*          agg    = (float*)ws;          ws += NH * sizeof(float);
  float*          cnt    = (float*)ws;          ws += (size_t)N * sizeof(float);
  unsigned short* hb     = (unsigned short*)ws; ws += NH * sizeof(unsigned short);
  unsigned short* tb     = (unsigned short*)ws; ws += NH * sizeof(unsigned short);
  unsigned short* WrelT  = (unsigned short*)ws; ws += (size_t)R * HDIM * HDIM * sizeof(unsigned short);
  int*            sorted = (int*)ws;            ws += (size_t)Epad * sizeof(int);
  int*            rmeta  = (int*)ws;            // 48 ints: counts / offsets / cursors

  hipMemsetAsync(agg, 0, NH * sizeof(float), stream);
  hipMemsetAsync(cnt, 0, (size_t)N * sizeof(float), stream);
  hipMemsetAsync(rmeta, 0, 48 * sizeof(int), stream);
  hipMemsetAsync(sorted, 0xFF, (size_t)Epad * sizeof(int), stream);  // -1 sentinels

  // one-time bf16/transposed relation weights
  k_convW<<<(R * HDIM * HDIM + 255) / 256, 256, 0, stream>>>(W_rel, WrelT,
                                                             R * HDIM * HDIM);
  // h = relu(x @ W_in + b_in)  -> f32 h (residual) + bf16 hb (gather source)
  k_gemm64<true, false, true, true><<<(N + 63) / 64, 256, 0, stream>>>(
      x, W_in, b_in, h, hb, N);
  // counting sort of edges by relation + in-degree counts
  k_hist<<<(E + 255) / 256, 256, 0, stream>>>(edge_type, edge_index, cnt, rmeta, E);
  k_scan<<<1, 1, 0, stream>>>(rmeta, R);
  k_scatter<<<(E + 255) / 256, 256, 0, stream>>>(edge_type, rmeta, sorted, E);
  // per-relation edge GEMM + scatter-add into agg
  k_edge_gemm<<<Epad / 64, 256, 0, stream>>>(sorted, edge_index, edge_type, hb,
                                             WrelT, b_rel, agg, Epad, E);
  // h2 = h + agg/clip(cnt,1)  -> bf16 (reuse hb)
  k_combine<<<(unsigned)((NH + 255) / 256), 256, 0, stream>>>(h, agg, cnt, hb, N);
  // t = relu(h2 @ W_o1 + b_o1)  -> bf16 tb
  k_gemm64<true, true, false, true><<<(N + 63) / 64, 256, 0, stream>>>(
      hb, W_o1, b_o1, (float*)nullptr, tb, N);
  // emb = t @ W_o2 + b_o2  -> f32 d_out
  k_gemm64<false, true, true, false><<<(N + 63) / 64, 256, 0, stream>>>(
      tb, W_o2, b_o2, (float*)d_out, (unsigned short*)nullptr, N);
  // pred = emb @ W_p + b_p  -> d_out after emb
  k_pred<<<(N + 7) / 8, 256, 0, stream>>>((const float*)d_out, W_p, b_p,
                                          (float*)d_out + NH, N);
}
